// SmoothQuantLinear_18107582120420
// MI455X (gfx1250) — 8-bit matrix operand order repaired and checked by reading, compiled, not run on hardware
//
#include <hip/hip_runtime.h>


typedef int v8i __attribute__((ext_vector_type(8)));

#define Mdim 8192
#define Kdim 4096
#define Ndim 16384
#define LDS_STRIDE 80

__global__ __launch_bounds__(256) void colmax_kernel(const float* __restrict__ src,
                                                     unsigned* __restrict__ dst,
                                                     int rows, int rows_per_block) {
    int col = blockIdx.x * 256 + threadIdx.x;
    int r0  = blockIdx.y * rows_per_block;
    int r1  = r0 + rows_per_block;
    if (r1 > rows) r1 = rows;
    float m = 0.0f;
    for (int r = r0; r < r1; ++r)
        m = fmaxf(m, fabsf(src[(size_t)r * Kdim + col]));
    atomicMax(&dst[col], __float_as_uint(m));
}

__global__ __launch_bounds__(256) void smooth_kernel(const unsigned* __restrict__ im,
                                                     const unsigned* __restrict__ wm,
                                                     float* __restrict__ smooth,
                                                     float* __restrict__ rsmooth) {
    int k = blockIdx.x * 256 + threadIdx.x;
    float s = sqrtf(__uint_as_float(im[k]) * __uint_as_float(wm[k]));
    s = fmaxf(s, 1e-20f);
    smooth[k]  = s;
    rsmooth[k] = 1.0f / s;
}

__global__ __launch_bounds__(256) void quant_kernel(const float* __restrict__ src,
                                                    const float* __restrict__ colmul,
                                                    signed char* __restrict__ dst,
                                                    float* __restrict__ rowscale) {
    __shared__ float red[8];
    const int row  = blockIdx.x;
    const int tid  = threadIdx.x;
    const int base = tid * 16;
    const float* rp = src + (size_t)row * Kdim;

    float v[16];
    float amax = 0.0f;
#pragma unroll
    for (int j = 0; j < 4; ++j) {
        float4 a = *(const float4*)(rp + base + j * 4);
        float4 c = *(const float4*)(colmul + base + j * 4);
        v[j * 4 + 0] = a.x * c.x;
        v[j * 4 + 1] = a.y * c.y;
        v[j * 4 + 2] = a.z * c.z;
        v[j * 4 + 3] = a.w * c.w;
        amax = fmaxf(amax, fmaxf(fmaxf(fabsf(v[j * 4 + 0]), fabsf(v[j * 4 + 1])),
                                 fmaxf(fabsf(v[j * 4 + 2]), fabsf(v[j * 4 + 3]))));
    }
#pragma unroll
    for (int off = 16; off > 0; off >>= 1)
        amax = fmaxf(amax, __shfl_xor(amax, off, 32));
    if ((tid & 31) == 0) red[tid >> 5] = amax;
    __syncthreads();
    amax = red[0];
#pragma unroll
    for (int i = 1; i < 8; ++i) amax = fmaxf(amax, red[i]);

    float scale = exp2f(ceilf(log2f(fmaxf(amax * (1.0f / 127.0f), 1e-30f))));
    float inv   = 1.0f / scale;

    int wrd[4];
#pragma unroll
    for (int j = 0; j < 4; ++j) {
        int q[4];
#pragma unroll
        for (int e = 0; e < 4; ++e) {
            int qi = (int)rintf(v[j * 4 + e] * inv);
            qi = qi > 127 ? 127 : (qi < -127 ? -127 : qi);
            q[e] = qi;
        }
        wrd[j] = (q[0] & 255) | ((q[1] & 255) << 8) | ((q[2] & 255) << 16) | ((q[3] & 255) << 24);
    }
    int4 o = make_int4(wrd[0], wrd[1], wrd[2], wrd[3]);
    *(int4*)(dst + (size_t)row * Kdim + base) = o;
    if (tid == 0) rowscale[row] = scale;
}

__global__ __launch_bounds__(256) void gemm_kernel(const signed char* __restrict__ Aq,
                                                   const signed char* __restrict__ Bq,
                                                   const float* __restrict__ xscale,
                                                   const float* __restrict__ wscale,
                                                   const float* __restrict__ bias,
                                                   float* __restrict__ out) {
    __shared__ __align__(16) signed char As[128 * LDS_STRIDE];
    __shared__ __align__(16) signed char Bs[128 * LDS_STRIDE];

    const int tid  = threadIdx.x;
    const int lane = tid & 31;
    const int wv   = tid >> 5;
    const int wm   = wv >> 2;
    const int wn   = wv & 3;
    const int m16  = lane & 15;
    const int half = lane >> 4;

    const int rowBase = blockIdx.y * 128;
    const int colBase = blockIdx.x * 128;

    const v8i vzero = {0, 0, 0, 0, 0, 0, 0, 0};
    v8i acc[4][2];
#pragma unroll
    for (int a = 0; a < 4; ++a)
#pragma unroll
        for (int b = 0; b < 2; ++b) acc[a][b] = vzero;

    for (int kb = 0; kb < Kdim; kb += 64) {
#pragma unroll
        for (int i = 0; i < 2; ++i) {
            int id = tid + i * 256;
            int r  = id >> 2;
            int c  = (id & 3) * 16;
            *(int4*)&As[r * LDS_STRIDE + c] =
                *(const int4*)&Aq[(size_t)(rowBase + r) * Kdim + kb + c];
            *(int4*)&Bs[r * LDS_STRIDE + c] =
                *(const int4*)&Bq[(size_t)(colBase + r) * Kdim + kb + c];
        }
        __syncthreads();

        v8i afrag[4];
#pragma unroll
        for (int tm = 0; tm < 4; ++tm) {
            const signed char* p = &As[(wm * 64 + tm * 16 + m16) * LDS_STRIDE];
            int4 d0 = *(const int4*)(p + half * 16);
            int4 d1 = *(const int4*)(p + 32 + half * 16);
            v8i a = {d0.x, d0.y, d0.z, d0.w, d1.x, d1.y, d1.z, d1.w};
            afrag[tm] = a;
        }
        v8i bfrag[2];
#pragma unroll
        for (int tn = 0; tn < 2; ++tn) {
            const signed char* p = &Bs[(wn * 32 + tn * 16 + m16) * LDS_STRIDE];
            int4 lo = *(const int4*)(p + half * 16);
            int4 hi = *(const int4*)(p + 32 + half * 16);
            v8i b = {lo.x, lo.y, lo.z, lo.w, hi.x, hi.y, hi.z, hi.w};
            bfrag[tn] = b;
        }

        __builtin_amdgcn_sched_barrier(0);

#pragma unroll
        for (int tm = 0; tm < 4; ++tm)
#pragma unroll
            for (int tn = 0; tn < 2; ++tn)
                acc[tm][tn] = __builtin_amdgcn_wmma_i32_16x16x64_iu8(
                     true, afrag[tm],  true, bfrag[tn],
                    acc[tm][tn],  false,  false);
        __syncthreads();
    }

#pragma unroll
    for (int tm = 0; tm < 4; ++tm) {
#pragma unroll
        for (int tn = 0; tn < 2; ++tn) {
            int n = colBase + wn * 32 + tn * 16 + m16;
            float wsv = wscale[n];
            float bv  = bias[n];
#pragma unroll
            for (int r = 0; r < 8; ++r) {
                int m = rowBase + wm * 64 + tm * 16 + half * 8 + r;
                float xsv = xscale[m];
                out[(size_t)m * Ndim + n] = (float)acc[tm][tn][r] * xsv * wsv + bv;
            }
        }
    }
}

extern "C" void kernel_launch(void* const* d_in, const int* in_sizes, int n_in,
                              void* d_out, int out_size, void* d_ws, size_t ws_size,
                              hipStream_t stream) {
    const float* x    = (const float*)d_in[0];
    const float* w    = (const float*)d_in[1];
    const float* bias = (const float*)d_in[2];
    float* out        = (float*)d_out;

    unsigned* im_bits  = (unsigned*)d_ws;
    unsigned* wm_bits  = im_bits + Kdim;
    float*    smooth   = (float*)(wm_bits + Kdim);
    float*    rsmooth  = smooth + Kdim;
    float*    xscale   = rsmooth + Kdim;
    float*    wscale   = xscale + Mdim;
    signed char* xq    = (signed char*)(wscale + Ndim);
    signed char* wq    = xq + (size_t)Mdim * Kdim;

    hipMemsetAsync(im_bits, 0, 2 * Kdim * sizeof(unsigned), stream);

    colmax_kernel<<<dim3(Kdim / 256, Mdim / 256), 256, 0, stream>>>(x, im_bits, Mdim, 256);
    colmax_kernel<<<dim3(Kdim / 256, Ndim / 256), 256, 0, stream>>>(w, wm_bits, Ndim, 256);

    smooth_kernel<<<Kdim / 256, 256, 0, stream>>>(im_bits, wm_bits, smooth, rsmooth);

    quant_kernel<<<Mdim, 256, 0, stream>>>(x, rsmooth, xq, xscale);
    quant_kernel<<<Ndim, 256, 0, stream>>>(w, smooth, wq, wscale);

    gemm_kernel<<<dim3(Ndim / 128, Mdim / 128), 256, 0, stream>>>(xq, wq, xscale, wscale, bias, out);
}
